// EncoderLayer_78735340471043
// MI455X (gfx1250) — compile-verified, matrix operand order checked by reading, not run on hardware
//
#include <hip/hip_runtime.h>
#include <stdint.h>

// ---------------- problem constants ----------------
constexpr int kB  = 2;
constexpr int kS  = 2048;
constexpr int kD  = 1024;
constexpr int kH  = 16;
constexpr int kDK = 64;
constexpr int kE  = 8;
constexpr int kF  = 4096;
constexpr int kT  = kB * kS;            // 4096 tokens

typedef _Float16 half_t;
typedef __attribute__((ext_vector_type(16))) half_t v16h;
typedef __attribute__((ext_vector_type(8)))  half_t v8h;
typedef __attribute__((ext_vector_type(4)))  half_t v4h;
typedef __attribute__((ext_vector_type(8)))  float  v8f;

static __device__ __forceinline__ v8f wmma_f16(const v16h& a, const v16h& b, const v8f& c) {
  return __builtin_amdgcn_wmma_f32_16x16x32_f16(false, a, false, b, (short)0, c, false, false);
}
static __device__ __forceinline__ v16h join16(const v8h& lo, const v8h& hi) {
  return __builtin_shufflevector(lo, hi, 0, 1, 2, 3, 4, 5, 6, 7, 8, 9, 10, 11, 12, 13, 14, 15);
}

// ---------------- workspace layout (bytes) ----------------
constexpr size_t OFF_XH   = 0;                                   // [T,D] f16  LN1 out
constexpr size_t OFF_WQKV = OFF_XH   + (size_t)kT * kD * 2;      // [3D,D] f16
constexpr size_t OFF_QKV  = OFF_WQKV + (size_t)3 * kD * kD * 2;  // [T,3D] f16 (q,k used)
constexpr size_t OFF_VT   = OFF_QKV  + (size_t)kT * 3 * kD * 2;  // [B,H,DK,S] f16 (v transposed)
constexpr size_t OFF_WOUT = OFF_VT   + (size_t)kT * kD * 2;      // [D,D] f16
constexpr size_t OFF_CTX  = OFF_WOUT + (size_t)kD * kD * 2;      // [T,D] f16
constexpr size_t OFF_SRC1 = OFF_CTX  + (size_t)kT * kD * 2;      // [T,D] f32
constexpr size_t OFF_YH   = OFF_SRC1 + (size_t)kT * kD * 4;      // [T,D] f16  LN2 out
constexpr size_t OFF_GATE = OFF_YH   + (size_t)kT * kD * 2;      // [T]   f32
constexpr size_t OFF_CNT  = OFF_GATE + (size_t)kT * 4;           // [E]   i32
constexpr size_t OFF_ETOK = OFF_CNT  + 256;                      // [E,T] i32
constexpr size_t OFF_W1   = OFF_ETOK + (size_t)kE * kT * 4;      // [E,F,D] f16 (transposed)
constexpr size_t OFF_W2   = OFF_W1   + (size_t)kE * kD * kF * 2; // [E,D,F] f16 (transposed)
constexpr size_t OFF_H    = OFF_W2   + (size_t)kE * kF * kD * 2; // [T,F] f16

// ---------------- helper kernels ----------------
__global__ void convert_f32_f16_kernel(const float* __restrict__ in,
                                       half_t* __restrict__ out, int n) {
  int i = blockIdx.x * blockDim.x + threadIdx.x;
  int stride = gridDim.x * blockDim.x;
  for (; i < n; i += stride) out[i] = (half_t)in[i];
}

// out[n*K + k] (f16) = in[k*N + n] (f32); per-slice (blockIdx.z) matrices of K x N.
__global__ __launch_bounds__(256)
void convert_transpose_kernel(const float* __restrict__ in_base,
                              half_t* __restrict__ out_base, int K, int N) {
  __shared__ __align__(16) float tile[64][68];
  const float* in = in_base + (size_t)blockIdx.z * K * N;
  half_t* out = out_base + (size_t)blockIdx.z * K * N;
  int k0 = blockIdx.y * 64;
  int n0 = blockIdx.x * 64;
  int r  = threadIdx.x >> 2;          // 0..63
  int c  = (threadIdx.x & 3) * 16;    // 0,16,32,48
  const float* ip = in + (size_t)(k0 + r) * N + n0 + c;
  #pragma unroll
  for (int q = 0; q < 4; q++)
    *(float4*)&tile[r][c + q * 4] = *(const float4*)(ip + q * 4);
  __syncthreads();
  half_t tmp[16];
  #pragma unroll
  for (int i = 0; i < 16; i++) tmp[i] = (half_t)tile[c + i][r];
  half_t* op = out + (size_t)(n0 + r) * K + k0 + c;
  *(v8h*)op       = *(v8h*)&tmp[0];
  *(v8h*)(op + 8) = *(v8h*)&tmp[8];
}

__global__ __launch_bounds__(256)
void layernorm_f16_kernel(const float* __restrict__ x, const float* __restrict__ g,
                          const float* __restrict__ b, half_t* __restrict__ out) {
  __shared__ float s1[256], s2[256];
  int row = blockIdx.x;
  const float* xr = x + (size_t)row * kD;
  int i0 = threadIdx.x * 4;
  float4 v = *(const float4*)(xr + i0);
  float a1 = v.x + v.y + v.z + v.w;
  float a2 = v.x * v.x + v.y * v.y + v.z * v.z + v.w * v.w;
  s1[threadIdx.x] = a1; s2[threadIdx.x] = a2; __syncthreads();
  for (int o = 128; o > 0; o >>= 1) {
    if (threadIdx.x < (unsigned)o) { s1[threadIdx.x] += s1[threadIdx.x + o]; s2[threadIdx.x] += s2[threadIdx.x + o]; }
    __syncthreads();
  }
  float mu = s1[0] * (1.f / kD);
  float var = s2[0] * (1.f / kD) - mu * mu;
  float rs = rsqrtf(var + 1e-5f);
  float4 gg = *(const float4*)(g + i0);
  float4 bb = *(const float4*)(b + i0);
  v4h o4;
  o4[0] = (half_t)((v.x - mu) * rs * gg.x + bb.x);
  o4[1] = (half_t)((v.y - mu) * rs * gg.y + bb.y);
  o4[2] = (half_t)((v.z - mu) * rs * gg.z + bb.z);
  o4[3] = (half_t)((v.w - mu) * rs * gg.w + bb.w);
  *(v4h*)(out + (size_t)row * kD + i0) = o4;
}

__global__ void zero_cnt_kernel(int* cnt) { if (threadIdx.x < kE) cnt[threadIdx.x] = 0; }

// ---------------- generic tiled WMMA GEMM ----------------
// C[M,N] = A[M,K] @ B^T + bias, B stored N-major [N,K] (row-major, ldb = K-ish).
// EPI 0: Ch = (half)(acc+bias); V third (ncol>=2D) scattered into vt [B,H,DK,S]
// EPI 1: Cf = residual + acc + bias
// EPI 2: Ch = (half)relu(acc+bias), row-gathered (MoE up)
// EPI 3: Cf = residual + gate[row]*(acc+bias), row-gathered (MoE down + combine)
constexpr int BM = 64, BN = 64, BK = 32;

template <int EPI>
__global__ __launch_bounds__(128)
void gemm_wmma_kernel(const half_t* __restrict__ A, int lda,
                      const half_t* __restrict__ Bm, int ldb,
                      const float* __restrict__ bias,
                      const float* __restrict__ residual,
                      const float* __restrict__ gatev,
                      const int* __restrict__ row_idx,
                      const int* __restrict__ row_cnt,
                      float* __restrict__ Cf, half_t* __restrict__ Ch,
                      half_t* __restrict__ vt,
                      int ldc, int M, int N, int K) {
  int Meff = M;
  if (row_cnt) Meff = *row_cnt;
  int mbase = blockIdx.y * BM;
  if (mbase >= Meff) return;
  int nbase = blockIdx.x * BN;

  __shared__ __align__(16) half_t As[BM][BK + 8];
  __shared__ __align__(16) half_t Bs[BN][BK + 8];   // N-major: frag reads contiguous in K

  int tid  = threadIdx.x;
  int wave = tid >> 5;
  int lane = tid & 31;
  int wm = (wave >> 1) * 32;
  int wn = (wave & 1) * 32;
  int lr = lane & 15;
  int kb = (lane >> 4) * 8;

  v8f c[2][2];
  for (int i = 0; i < 2; i++) for (int j = 0; j < 2; j++) c[i][j] = v8f{};

  int arow = tid >> 1;
  int acol = (tid & 1) * 16;
  int g_arow = -1;
  {
    int r = mbase + arow;
    if (r < Meff) g_arow = row_idx ? row_idx[r] : r;
  }
  const half_t* bsrc = Bm + (size_t)(nbase + arow) * ldb + acol;

  for (int k0 = 0; k0 < K; k0 += BK) {
    // ---- stage A (vector) ----
    if (g_arow >= 0) {
      const half_t* sp = A + (size_t)g_arow * lda + k0 + acol;
      *(v8h*)&As[arow][acol]     = *(const v8h*)sp;
      *(v8h*)&As[arow][acol + 8] = *(const v8h*)(sp + 8);
    } else {
      *(v8h*)&As[arow][acol]     = v8h{};
      *(v8h*)&As[arow][acol + 8] = v8h{};
    }
    // ---- stage B (vector, no transpose: B already N-major) ----
    *(v8h*)&Bs[arow][acol]     = *(const v8h*)(bsrc + k0);
    *(v8h*)&Bs[arow][acol + 8] = *(const v8h*)(bsrc + k0 + 8);

    if (k0 + BK < K) {   // prefetch next K tiles (global_prefetch_b8)
      if (g_arow >= 0) __builtin_prefetch(A + (size_t)g_arow * lda + k0 + BK + acol, 0, 1);
      __builtin_prefetch(bsrc + k0 + BK, 0, 1);
    }
    __syncthreads();

    // ---- fragments (2x v8h per frag) + WMMA ----
    v16h af[2], bf[2];
    #pragma unroll
    for (int i = 0; i < 2; i++) {
      int m = wm + i * 16 + lr;
      af[i] = join16(*(const v8h*)&As[m][kb], *(const v8h*)&As[m][16 + kb]);
    }
    #pragma unroll
    for (int j = 0; j < 2; j++) {
      int n = wn + j * 16 + lr;
      bf[j] = join16(*(const v8h*)&Bs[n][kb], *(const v8h*)&Bs[n][16 + kb]);
    }
    #pragma unroll
    for (int i = 0; i < 2; i++)
      #pragma unroll
      for (int j = 0; j < 2; j++)
        c[i][j] = wmma_f16(af[i], bf[j], c[i][j]);
    __syncthreads();
  }

  // ---- epilogue ----
  int lg = lane >> 4;
  #pragma unroll
  for (int i = 0; i < 2; i++) {
    #pragma unroll
    for (int j = 0; j < 2; j++) {
      #pragma unroll
      for (int r = 0; r < 8; r++) {
        int mrow = mbase + wm + i * 16 + r + lg * 8;
        int ncol = nbase + wn + j * 16 + lr;
        if (mrow >= Meff) continue;
        int grow = row_idx ? row_idx[mrow] : mrow;
        float v = c[i][j][r] + (bias ? bias[ncol] : 0.f);
        if (EPI == 0) {
          if (ncol >= 2 * kD) {   // V third -> transposed [b,h,d,s] layout
            int hh = (ncol - 2 * kD) >> 6;
            int dd = (ncol - 2 * kD) & 63;
            int bb = grow >> 11;          // /kS
            int ss = grow & (kS - 1);
            vt[((((size_t)bb * kH + hh) * kDK + dd) << 11) + ss] = (half_t)v;
          } else {
            Ch[(size_t)grow * ldc + ncol] = (half_t)v;
          }
        } else if (EPI == 1) {
          Cf[(size_t)grow * ldc + ncol] = residual[(size_t)grow * ldc + ncol] + v;
        } else if (EPI == 2) {
          Ch[(size_t)grow * ldc + ncol] = (half_t)fmaxf(v, 0.f);
        } else {
          Cf[(size_t)grow * ldc + ncol] =
              residual[(size_t)grow * ldc + ncol] + gatev[grow] * v;
        }
      }
    }
  }
}

// ---------------- flash attention ----------------
__global__ __launch_bounds__(128)
void attn_kernel(const half_t* __restrict__ qkv, const half_t* __restrict__ vt,
                 const uint8_t* __restrict__ pad, half_t* __restrict__ ctx) {
  constexpr int QT = 64, KT = 64, LD3 = 3 * kD;
  int qbase = blockIdx.x * QT;
  int h  = blockIdx.y;
  int bb = blockIdx.z;
  size_t row0 = (size_t)bb * kS;

  __shared__ __align__(16) half_t Ks[KT][kDK + 8];   // [key][d]
  __shared__ __align__(16) half_t Vs[kDK][KT + 8];   // [d][key] (from vt)
  __shared__ __align__(16) float  St[QT][68];
  __shared__ __align__(16) half_t Ps[QT][KT + 8];
  __shared__ float mrow[QT], lsumr[QT], alph[QT], kbias[KT];
  __shared__ float redm[QT][2], reds[QT][2];

  int tid = threadIdx.x, wave = tid >> 5, lane = tid & 31;
  int lr = lane & 15, lg = lane >> 4, kb = lg * 8;

  if (tid < QT) { mrow[tid] = -1e30f; lsumr[tid] = 0.f; }

  // Q fragments: wave's 16 rows, DK=64 -> 2 k-steps; vector global loads
  v16h aq[2];
  {
    int qrow = qbase + wave * 16 + lr;
    const half_t* qp = qkv + (row0 + qrow) * LD3 + h * kDK;
    #pragma unroll
    for (int ks = 0; ks < 2; ks++)
      aq[ks] = join16(*(const v8h*)(qp + ks * 32 + kb),
                      *(const v8h*)(qp + ks * 32 + 16 + kb));
  }

  v8f o[4];
  for (int j = 0; j < 4; j++) o[j] = v8f{};
  __syncthreads();

  for (int kc = 0; kc < kS; kc += KT) {
    // ---- stage K [key][d] and V [d][key] chunks (all vector) ----
    {
      int r  = tid >> 1;
      int hc = (tid & 1) * 32;
      const half_t* kp = qkv + (row0 + kc + r) * LD3 + kD + h * kDK + hc;
      const half_t* vp = vt + (((size_t)bb * kH + h) * kDK + r) * kS + kc + hc;
      #pragma unroll
      for (int q = 0; q < 4; q++) {
        *(v8h*)&Ks[r][hc + q * 8] = *(const v8h*)(kp + q * 8);
        *(v8h*)&Vs[r][hc + q * 8] = *(const v8h*)(vp + q * 8);
      }
      if (tid < KT) kbias[tid] = pad[(size_t)bb * kS + kc + tid] ? -1e9f : 0.f;
    }
    __syncthreads();

    // ---- scores: wave's 16 q-rows x 64 keys ----
    #pragma unroll
    for (int j = 0; j < 4; j++) {
      v8f s = v8f{};
      int key = j * 16 + lr;
      #pragma unroll
      for (int ks = 0; ks < 2; ks++) {
        v16h bk = join16(*(const v8h*)&Ks[key][ks * 32 + kb],
                         *(const v8h*)&Ks[key][ks * 32 + 16 + kb]);
        s = wmma_f16(aq[ks], bk, s);
      }
      #pragma unroll
      for (int r = 0; r < 8; r++) {
        int m = wave * 16 + r + lg * 8;
        St[m][key] = s[r] * 0.125f + kbias[key];
      }
    }
    __syncthreads();

    // ---- online softmax: 2 threads per row, vectorized ----
    {
      int rrow = tid >> 1, hf = tid & 1;
      int base = hf * 32;
      float lmax = -1e30f;
      #pragma unroll
      for (int q = 0; q < 8; q++) {
        float4 vv = *(const float4*)&St[rrow][base + q * 4];
        lmax = fmaxf(lmax, fmaxf(fmaxf(vv.x, vv.y), fmaxf(vv.z, vv.w)));
      }
      redm[rrow][hf] = lmax;
      __syncthreads();
      float newm = fmaxf(mrow[rrow], fmaxf(redm[rrow][0], redm[rrow][1]));
      float lsum = 0.f;
      #pragma unroll
      for (int g = 0; g < 4; g++) {
        float4 a = *(const float4*)&St[rrow][base + g * 8];
        float4 b = *(const float4*)&St[rrow][base + g * 8 + 4];
        float p0 = __expf(a.x - newm), p1 = __expf(a.y - newm);
        float p2 = __expf(a.z - newm), p3 = __expf(a.w - newm);
        float p4 = __expf(b.x - newm), p5 = __expf(b.y - newm);
        float p6 = __expf(b.z - newm), p7 = __expf(b.w - newm);
        v8h pk;
        pk[0] = (half_t)p0; pk[1] = (half_t)p1; pk[2] = (half_t)p2; pk[3] = (half_t)p3;
        pk[4] = (half_t)p4; pk[5] = (half_t)p5; pk[6] = (half_t)p6; pk[7] = (half_t)p7;
        *(v8h*)&Ps[rrow][base + g * 8] = pk;
        lsum += p0 + p1 + p2 + p3 + p4 + p5 + p6 + p7;
      }
      reds[rrow][hf] = lsum;
      __syncthreads();
      if (hf == 0) {
        float a = __expf(mrow[rrow] - newm);
        alph[rrow]  = a;
        lsumr[rrow] = lsumr[rrow] * a + reds[rrow][0] + reds[rrow][1];
        mrow[rrow]  = newm;
      }
    }
    __syncthreads();

    // ---- O = O*alpha + P @ V ----
    v16h ap[2];
    {
      int prow = wave * 16 + lr;
      #pragma unroll
      for (int ks = 0; ks < 2; ks++)
        ap[ks] = join16(*(const v8h*)&Ps[prow][ks * 32 + kb],
                        *(const v8h*)&Ps[prow][ks * 32 + 16 + kb]);
    }
    #pragma unroll
    for (int j = 0; j < 4; j++) {
      #pragma unroll
      for (int r = 0; r < 8; r++) o[j][r] *= alph[wave * 16 + r + lg * 8];
      int d = j * 16 + lr;
      #pragma unroll
      for (int ks = 0; ks < 2; ks++) {
        v16h bv = join16(*(const v8h*)&Vs[d][ks * 32 + kb],
                         *(const v8h*)&Vs[d][ks * 32 + 16 + kb]);
        o[j] = wmma_f16(ap[ks], bv, o[j]);
      }
    }
    __syncthreads();
  }

  // ---- finalize ----
  #pragma unroll
  for (int j = 0; j < 4; j++) {
    #pragma unroll
    for (int r = 0; r < 8; r++) {
      int m = wave * 16 + r + lg * 8;
      int d = j * 16 + lr;
      ctx[(row0 + qbase + m) * kD + h * kDK + d] = (half_t)(o[j][r] / lsumr[m]);
    }
  }
}

// ---------------- router ----------------
__global__ __launch_bounds__(128)
void router_kernel(const half_t* __restrict__ y, const float* __restrict__ rw,
                   const float* __restrict__ rb, const uint8_t* __restrict__ tmask,
                   float* __restrict__ gate, int* __restrict__ cnt, int* __restrict__ etok) {
  int t = blockIdx.x;
  int i0 = threadIdx.x * 8;             // 128 threads x 8 = D
  v8h yv = *(const v8h*)(y + (size_t)t * kD + i0);
  float yf[8];
  #pragma unroll
  for (int i = 0; i < 8; i++) yf[i] = (float)yv[i];
  float acc[kE];
  #pragma unroll
  for (int e = 0; e < kE; e++) {
    float4 wa = *(const float4*)(rw + e * kD + i0);
    float4 wb = *(const float4*)(rw + e * kD + i0 + 4);
    acc[e] = yf[0] * wa.x + yf[1] * wa.y + yf[2] * wa.z + yf[3] * wa.w +
             yf[4] * wb.x + yf[5] * wb.y + yf[6] * wb.z + yf[7] * wb.w;
  }
  __shared__ float part[128];
  __shared__ float logits[kE];
  for (int e = 0; e < kE; e++) {
    part[threadIdx.x] = acc[e]; __syncthreads();
    for (int o = 64; o > 0; o >>= 1) {
      if (threadIdx.x < (unsigned)o) part[threadIdx.x] += part[threadIdx.x + o];
      __syncthreads();
    }
    if (threadIdx.x == 0) logits[e] = part[0] + rb[e];
    __syncthreads();
  }
  if (threadIdx.x == 0) {
    float mx = -1e30f; int mi = 0;
    #pragma unroll
    for (int e = 0; e < kE; e++) if (logits[e] > mx) { mx = logits[e]; mi = e; }
    float sum = 0.f;
    #pragma unroll
    for (int e = 0; e < kE; e++) sum += __expf(logits[e] - mx);
    float g = 1.f / sum;                      // top-1 softmax prob
    g *= tmask[t] ? 1.f : 0.f;
    gate[t] = g;
    int slot = atomicAdd(&cnt[mi], 1);
    etok[mi * kT + slot] = t;
  }
}

// ---------------- launch ----------------
extern "C" void kernel_launch(void* const* d_in, const int* in_sizes, int n_in,
                              void* d_out, int out_size, void* d_ws, size_t ws_size,
                              hipStream_t stream) {
  (void)in_sizes; (void)n_in; (void)out_size; (void)ws_size;
  const float*   src   = (const float*)d_in[0];
  const uint8_t* pad   = (const uint8_t*)d_in[1];
  const uint8_t* tmask = (const uint8_t*)d_in[2];
  const float*   ln1g  = (const float*)d_in[3];
  const float*   ln1b  = (const float*)d_in[4];
  const float*   wqkv  = (const float*)d_in[5];
  const float*   bqkv  = (const float*)d_in[6];
  const float*   wout  = (const float*)d_in[7];
  const float*   bout  = (const float*)d_in[8];
  const float*   ln2g  = (const float*)d_in[9];
  const float*   ln2b  = (const float*)d_in[10];
  const float*   rw    = (const float*)d_in[11];
  const float*   rb    = (const float*)d_in[12];
  const float*   w1    = (const float*)d_in[13];
  const float*   b1    = (const float*)d_in[14];
  const float*   w2    = (const float*)d_in[15];
  const float*   b2    = (const float*)d_in[16];
  float* out = (float*)d_out;

  char* ws = (char*)d_ws;
  half_t* xh    = (half_t*)(ws + OFF_XH);
  half_t* wqkvh = (half_t*)(ws + OFF_WQKV);
  half_t* qkvh  = (half_t*)(ws + OFF_QKV);
  half_t* vt    = (half_t*)(ws + OFF_VT);
  half_t* wouth = (half_t*)(ws + OFF_WOUT);
  half_t* ctxh  = (half_t*)(ws + OFF_CTX);
  float*  src1  = (float*)(ws + OFF_SRC1);
  half_t* yh    = (half_t*)(ws + OFF_YH);
  float*  gate  = (float*)(ws + OFF_GATE);
  int*    cnt   = (int*)(ws + OFF_CNT);
  int*    etok  = (int*)(ws + OFF_ETOK);
  half_t* w1t   = (half_t*)(ws + OFF_W1);   // [E][F][D]
  half_t* w2t   = (half_t*)(ws + OFF_W2);   // [E][D][F]
  half_t* hbuf  = (half_t*)(ws + OFF_H);

  // 1) weight conversion (w1/w2 transposed to N-major)
  convert_f32_f16_kernel<<<2048, 256, 0, stream>>>(wqkv, wqkvh, 3 * kD * kD);
  convert_f32_f16_kernel<<<1024, 256, 0, stream>>>(wout, wouth, kD * kD);
  convert_transpose_kernel<<<dim3(kF / 64, kD / 64, kE), 256, 0, stream>>>(w1, w1t, kD, kF);
  convert_transpose_kernel<<<dim3(kD / 64, kF / 64, kE), 256, 0, stream>>>(w2, w2t, kF, kD);

  // 2) LN1
  layernorm_f16_kernel<<<kT, 256, 0, stream>>>(src, ln1g, ln1b, xh);

  // 3) QKV projection (V third scattered into vt)
  gemm_wmma_kernel<0><<<dim3(3 * kD / BN, kT / BM), 128, 0, stream>>>(
      xh, kD, wqkvh, kD, bqkv, nullptr, nullptr, nullptr, nullptr,
      nullptr, qkvh, vt, 3 * kD, kT, 3 * kD, kD);

  // 4) attention
  attn_kernel<<<dim3(kS / 64, kH, kB), 128, 0, stream>>>(qkvh, vt, pad, ctxh);

  // 5) out projection + residual -> src1
  gemm_wmma_kernel<1><<<dim3(kD / BN, kT / BM), 128, 0, stream>>>(
      ctxh, kD, wouth, kD, bout, src, nullptr, nullptr, nullptr,
      src1, nullptr, nullptr, kD, kT, kD, kD);

  // 6) LN2, routing
  layernorm_f16_kernel<<<kT, 256, 0, stream>>>(src1, ln2g, ln2b, yh);
  zero_cnt_kernel<<<1, 32, 0, stream>>>(cnt);
  router_kernel<<<kT, 128, 0, stream>>>(yh, rw, rb, tmask, gate, cnt, etok);

  // 7) MoE expert GEMMs (row gather/scatter; blocks past count early-exit)
  for (int e = 0; e < kE; e++) {
    gemm_wmma_kernel<2><<<dim3(kF / BN, kT / BM), 128, 0, stream>>>(
        yh, kD, w1t + (size_t)e * kD * kF, kD, b1 + e * kF,
        nullptr, nullptr, etok + e * kT, cnt + e,
        nullptr, hbuf, nullptr, kF, kT, kF, kD);
  }
  for (int e = 0; e < kE; e++) {
    gemm_wmma_kernel<3><<<dim3(kD / BN, kT / BM), 128, 0, stream>>>(
        hbuf, kF, w2t + (size_t)e * kF * kD, kF, b2 + e * kD,
        src1, gate, etok + e * kT, cnt + e,
        out, nullptr, nullptr, kD, kT, kD, kF);
  }
}
